// S4Layer_87634512708175
// MI455X (gfx1250) — matrix operand order repaired and checked by reading, not compiled, not run on hardware
//
#include <hip/hip_runtime.h>
#include <hip/hip_bf16.h>

// ---- sizes ----
#define D_MODEL 256
#define N_STATE 64
#define L_MAX   2048
#define BATCH   16

typedef _Float16 v16h __attribute__((ext_vector_type(16)));
typedef _Float16 v8h  __attribute__((ext_vector_type(8)));
typedef _Float16 v4h  __attribute__((ext_vector_type(4)));
typedef float    v8f  __attribute__((ext_vector_type(8)));
typedef unsigned int u32x4 __attribute__((ext_vector_type(4)));
typedef int          i32x8 __attribute__((ext_vector_type(8)));
typedef int          i32x4 __attribute__((ext_vector_type(4)));

union F16x16 { v16h v; v8h h[2]; _Float16 e[16]; };

// ============================================================================
// Kernel 1: Cauchy kernel at roots of unity.
// at_roots[d,l] = c_l * (k00 - k01*k10/(1+k11)),  g = (2/step)(1-w)/(1+w)
// grid (L/256, D), block 256
// ============================================================================
__global__ void s4_cauchy(const float* __restrict__ p_ri,
                          const float* __restrict__ lam_ri,
                          const float* __restrict__ B_ri,
                          const float* __restrict__ Ct_ri,
                          const float* __restrict__ log_step,
                          float2* __restrict__ at) {
  const int d = blockIdx.y;
  const int l = blockIdx.x * 256 + threadIdx.x;
  const float step = __expf(log_step[d]);
  const float ang = 6.28318530717958647692f * (float)l / (float)L_MAX;
  float wi, wr;
  __sincosf(ang, &wi, &wr);                 // omega = (wr, wi)
  const float dr = 1.f + wr, di = wi;       // 1 + omega
  const float nr = 1.f - wr, ni = -wi;      // 1 - omega
  const float d2 = dr * dr + di * di;
  const float tr = (nr * dr + ni * di) / d2;
  const float ti = (ni * dr - nr * di) / d2;
  const float s2 = 2.f / step;
  const float gr = s2 * tr, gi = s2 * ti;   // g
  const float cr = 2.f * dr / d2, ci = -2.f * di / d2;  // c = 2/(1+omega)

  float k00r = 0.f, k00i = 0.f, k01r = 0.f, k01i = 0.f;
  float k10r = 0.f, k10i = 0.f, k11r = 0.f, k11i = 0.f;
#pragma unroll 4
  for (int n = 0; n < N_STATE; ++n) {
    const float pr = p_ri[2 * n], pi = p_ri[2 * n + 1];
    const float lr = lam_ri[2 * n], li = lam_ri[2 * n + 1];
    const float br = B_ri[(d * N_STATE + n) * 2], bi = B_ri[(d * N_STATE + n) * 2 + 1];
    const float ar = Ct_ri[(d * N_STATE + n) * 2], ai = Ct_ri[(d * N_STATE + n) * 2 + 1];
    const float er = gr - lr, ei = gi - li;
    const float inv2 = 1.f / (er * er + ei * ei);
    const float ir = er * inv2, ii = -ei * inv2;  // 1/(g - lam)
    const float w00r = ar * br + ai * bi, w00i = ar * bi - ai * br;
    const float w01r = ar * pr + ai * pi, w01i = ar * pi - ai * pr;
    const float w10r = pr * br + pi * bi, w10i = pr * bi - pi * br;
    const float w11r = pr * pr + pi * pi;
    k00r += w00r * ir - w00i * ii; k00i += w00r * ii + w00i * ir;
    k01r += w01r * ir - w01i * ii; k01i += w01r * ii + w01i * ir;
    k10r += w10r * ir - w10i * ii; k10i += w10r * ii + w10i * ir;
    k11r += w11r * ir;             k11i += w11r * ii;
  }
  const float q1r = k01r * k10r - k01i * k10i;
  const float q1i = k01r * k10i + k01i * k10r;
  const float hr = 1.f + k11r, hi = k11i;
  const float h2i = 1.f / (hr * hr + hi * hi);
  const float qr = (q1r * hr + q1i * hi) * h2i;
  const float qi = (q1i * hr - q1r * hi) * h2i;
  const float vr = k00r - qr, vi = k00i - qi;
  at[d * L_MAX + l] = make_float2(cr * vr - ci * vi, cr * vi + ci * vr);
}

// ============================================================================
// Kernel 2: K[d,l] = (1/L) Re( FFT_L(at_roots[d,:])[l] )   (== ifft + flip)
// Stockham radix-2 autosort in LDS. One workgroup (256 thr) per channel.
// ============================================================================
__global__ void s4_fftK(const float2* __restrict__ at, _Float16* __restrict__ Kh) {
  __shared__ float2 bufA[L_MAX];
  __shared__ float2 bufB[L_MAX];
  const int d = blockIdx.x;
  const int tid = threadIdx.x;
  for (int i = tid; i < L_MAX; i += 256) bufA[i] = at[d * L_MAX + i];
  __syncthreads();
  float2* src = bufA;
  float2* dst = bufB;
  int l = L_MAX / 2, m = 1;
  for (int s = 0; s < 11; ++s) {
    for (int i = tid; i < L_MAX / 2; i += 256) {
      const int j = i / m;
      const int k = i - j * m;
      const float angw = -3.14159265358979323846f * (float)j / (float)l;
      float swi, swr;
      __sincosf(angw, &swi, &swr);
      const float2 a = src[k + j * m];
      const float2 b = src[k + j * m + l * m];
      dst[k + 2 * j * m] = make_float2(a.x + b.x, a.y + b.y);
      const float trr = a.x - b.x, tii = a.y - b.y;
      dst[k + 2 * j * m + m] = make_float2(trr * swr - tii * swi, trr * swi + tii * swr);
    }
    __syncthreads();
    float2* tmp = src; src = dst; dst = tmp;
    l >>= 1; m <<= 1;
  }
  for (int i = tid; i < L_MAX; i += 256)
    Kh[d * L_MAX + i] = (_Float16)(src[i].x * (1.0f / (float)L_MAX));
}

// ============================================================================
// Kernel 3: u (B,L,D) f32  ->  u_h[d][t][b] f16 (coalesced both sides via LDS)
// grid (L/16, D/64), block 256.  LDS tile 64x16x16 f16 = 32 KB.
// ============================================================================
__global__ void s4_transpose_u(const float* __restrict__ u, _Float16* __restrict__ u_h) {
  __shared__ _Float16 tile[64 * 16 * 16];
  const int t0 = blockIdx.x * 16;
  const int d0 = blockIdx.y * 64;
  const int tid = threadIdx.x;
#pragma unroll
  for (int it = 0; it < 64; ++it) {
    const int flat = tid + it * 256;          // 16384 elements
    const int dd = flat & 63;
    const int tt = (flat >> 6) & 15;
    const int b = flat >> 10;
    const float v = u[((size_t)b * L_MAX + (t0 + tt)) * D_MODEL + (d0 + dd)];
    tile[(dd * 16 + tt) * 16 + b] = (_Float16)v;
  }
  __syncthreads();
#pragma unroll
  for (int it = 0; it < 64; ++it) {
    const int flat = tid + it * 256;
    const int b = flat & 15;
    const int tt = (flat >> 4) & 15;
    const int dd = flat >> 8;
    u_h[((size_t)(d0 + dd) * L_MAX + (t0 + tt)) * BATCH + b] = tile[(dd * 16 + tt) * 16 + b];
  }
}

// ============================================================================
// TDM 1-D copy: 512 f16 (1 KB) contiguous global -> LDS, async (TENSORcnt).
// D# per ISA cdna5 §8.3/8.4: group0 = count|lds_addr|global_addr|type=2,
// group1 = data_size=2B, tensor_dim0=tile_dim0=512, dims1=1, stride0=512.
// This toolchain exposes the 6-arg builtin: (g0, g1, g2, g3, g_extra, cpol).
// ============================================================================
__device__ __forceinline__ void tdm_load_u_tile(unsigned lds_off, const void* gptr) {
  const unsigned long long ga = (unsigned long long)(uintptr_t)gptr;
  u32x4 g0;
  g0[0] = 1u;                                        // count=1, user descriptor
  g0[1] = lds_off;                                   // LDS byte address
  g0[2] = (unsigned)(ga & 0xffffffffu);              // global_addr[31:0]
  g0[3] = (unsigned)((ga >> 32) & 0x1ffffffu) | (2u << 30);  // addr[56:32] | type=2
  i32x8 g1;
  g1[0] = 0x00010000;        // workgroup_mask=0, data_size=1 (2 bytes)
  g1[1] = (int)(512u << 16); // tensor_dim0[15:0] = 512 (in elements)
  g1[2] = (int)(1u << 16);   // tensor_dim0[31:16]=0, tensor_dim1[15:0]=1
  g1[3] = (int)(512u << 16); // tensor_dim1[31:16]=0, tile_dim0=512
  g1[4] = 1;                 // tile_dim1=1, tile_dim2=0
  g1[5] = 512;               // tensor_dim0_stride[31:0]
  g1[6] = 0;                 // stride0[47:32]=0, tensor_dim1_stride[15:0]=0
  g1[7] = 0;
  i32x4 z4 = {0, 0, 0, 0};              // groups 2/3 unused (2-D descriptor)
  i32x8 z8 = {0, 0, 0, 0, 0, 0, 0, 0};  // trailing group (6-arg builtin form)
  __builtin_amdgcn_tensor_load_to_lds(g0, g1, z4, z4, z8, 0);
}

// ============================================================================
// Kernel 4: causal Toeplitz matmul per channel via V_WMMA_F32_16X16X32_F16,
// with TDM double-buffered async staging of the u tiles.
// WG = (row block of 256, channel d), 128 threads = 4 waves, 4 tiles/wave.
// A[m,k] = K[d, (l0+m)-(t0+k)] (0 above diagonal) — read from reversed-K LDS.
// B[k,n] = u_h[d, t0+k, n] (n = batch).
// Dynamic LDS layout (base offset 0 — only dynamic smem in this kernel):
//   [0,4352)      Rev   : reversed K, zero-padded          (needs <= 4160 B)
//   [4352,5376)   Ust0  : raw u tile ping  [k][n], 1 KB    (TDM destination)
//   [5376,6400)   Ust1  : raw u tile pong  [k][n], 1 KB    (TDM destination)
//   [6400,7424)   Bst   : transposed tile  [n][k], 1 KB
// grid (L/256, D), block 128, smem 7424 B.
// ============================================================================
#define SM_REV  0u
#define SM_UST0 4352u
#define SM_UST1 5376u
#define SM_BST  6400u
#define SM_SIZE 7424u

__global__ void __launch_bounds__(128)
s4_conv_wmma(const _Float16* __restrict__ Kh,
             const _Float16* __restrict__ u_h,
             const float* __restrict__ u,
             const float* __restrict__ Dp,
             float* __restrict__ out) {
  extern __shared__ char smem[];
  _Float16* Rev  = (_Float16*)(smem + SM_REV);
  _Float16* Ust0 = (_Float16*)(smem + SM_UST0);
  _Float16* Ust1 = (_Float16*)(smem + SM_UST1);
  _Float16* Bst  = (_Float16*)(smem + SM_BST);

  const int r0 = blockIdx.x * 256;
  const int d = blockIdx.y;
  const int Kext = r0 + 256;
  const int nchunks = Kext >> 5;
  const int tid = threadIdx.x;
  const int lane = tid & 31;
  const int w = tid >> 5;
  const int nn = lane & 15;                // A row m / B col n / D col n
  const int g = lane >> 4;                 // lane group

  const _Float16* u_row = u_h + (size_t)d * L_MAX * BATCH;

  // kick off async TDM for chunk 0 while we stage the reversed kernel
  if (w == 0) tdm_load_u_tile(SM_UST0, u_row);

  // stage reversed kernel: Rev[j] = K[Kext-1-j], Rev[j>=Kext] = 0
  for (int j = tid; j < Kext + 32; j += 128)
    Rev[j] = (j < Kext) ? Kh[(size_t)d * L_MAX + (Kext - 1 - j)] : (_Float16)0.f;

  v8f acc[4] = {};
  const float dpv = Dp[d];

  for (int i = 0; i < nchunks; ++i) {
    const int t0 = i << 5;
    // wave 0: prefetch chunk i+1 via TDM, then drain so chunk i is resident
    if (w == 0) {
      if (i + 1 < nchunks) {
        tdm_load_u_tile((i & 1) ? SM_UST0 : SM_UST1,
                        u_row + (size_t)(t0 + 32) * BATCH);
        __builtin_amdgcn_s_wait_tensorcnt(1);   // chunk i done, i+1 in flight
      } else {
        __builtin_amdgcn_s_wait_tensorcnt(0);   // last chunk done
      }
    }
    __syncthreads();                            // Ust[i&1] visible to all waves

    // ---- LDS transpose: Ust [k][n] -> Bst [n][k] ----
    {
      const _Float16* Ust = (i & 1) ? Ust1 : Ust0;
      const int e0 = tid * 4;                   // 512 halfs / 128 threads
      const int k = e0 >> 4;
      const int n = e0 & 15;
      const v4h x = *(const v4h*)&Ust[k * 16 + n];
      Bst[(n + 0) * 32 + k] = x[0];
      Bst[(n + 1) * 32 + k] = x[1];
      Bst[(n + 2) * 32 + k] = x[2];
      Bst[(n + 3) * 32 + k] = x[3];
    }
    __syncthreads();

    // ---- B fragment: e<8 -> k=g*8+e, e>=8 -> k=16+g*8+(e-8) ----
    F16x16 bf;
    bf.h[0] = *(const v8h*)&Bst[nn * 32 + g * 8];
    bf.h[1] = *(const v8h*)&Bst[nn * 32 + 16 + g * 8];

#pragma unroll
    for (int tt = 0; tt < 4; ++tt) {
      const int l0 = r0 + w * 64 + tt * 16;
      if (t0 > l0 + 15) continue;          // wave-uniform: EXEC stays all-ones
      // A fragment, ISA 16-bit A layout: e<8 -> k=g*8+e, e>=8 -> k=16+g*8+(e-8)
      const int c = l0 + nn - t0;          // K index at k=0 (>=0 for active tiles)
      const int jb = (Kext - 1 - c) + g * 8;
      F16x16 af;
#pragma unroll
      for (int e = 0; e < 8; ++e) {
        af.e[e] = Rev[jb + e];             // zeros past diagonal via Rev padding
        af.e[8 + e] = Rev[jb + 16 + e];
      }
      acc[tt] = __builtin_amdgcn_wmma_f32_16x16x32_f16(
          false, af.v, false, bf.v, (short)0, acc[tt], false, false);
    }
    __syncthreads();                       // Bst reads done before next transpose
  }

  // ---- epilogue: D layout lane->n, vgpr r -> m = r + 8*g; fuse skip Dp*u ----
#pragma unroll
  for (int tt = 0; tt < 4; ++tt) {
    const int l0 = r0 + w * 64 + tt * 16;
#pragma unroll
    for (int r = 0; r < 8; ++r) {
      const int m = r + 8 * g;
      const size_t idx = ((size_t)nn * L_MAX + (l0 + m)) * D_MODEL + d;
      out[idx] = acc[tt][r] + dpv * u[idx];
    }
  }
}

// ============================================================================
extern "C" void kernel_launch(void* const* d_in, const int* in_sizes, int n_in,
                              void* d_out, int out_size, void* d_ws, size_t ws_size,
                              hipStream_t stream) {
  (void)in_sizes; (void)n_in; (void)out_size; (void)ws_size;
  const float* u        = (const float*)d_in[0];   // (16,2048,256)
  const float* p_ri     = (const float*)d_in[1];   // (64,2)
  const float* lam_ri   = (const float*)d_in[2];   // (64,2)
  const float* B_ri     = (const float*)d_in[3];   // (256,64,2)
  const float* Ct_ri    = (const float*)d_in[4];   // (256,64,2)
  const float* Dp       = (const float*)d_in[5];   // (256)
  const float* log_step = (const float*)d_in[6];   // (256)
  float* out = (float*)d_out;                      // (16,2048,256)

  char* ws = (char*)d_ws;
  float2*   at  = (float2*)ws;                                   // 4 MB
  _Float16* Kh  = (_Float16*)(ws + (size_t)4 * 1024 * 1024);     // 1 MB
  _Float16* u_h = (_Float16*)(ws + (size_t)5 * 1024 * 1024);     // 16 MB

  s4_cauchy<<<dim3(L_MAX / 256, D_MODEL), 256, 0, stream>>>(
      p_ri, lam_ri, B_ri, Ct_ri, log_step, at);
  s4_fftK<<<dim3(D_MODEL), 256, 0, stream>>>(at, Kh);
  s4_transpose_u<<<dim3(L_MAX / 16, D_MODEL / 64), 256, 0, stream>>>(u, u_h);
  s4_conv_wmma<<<dim3(L_MAX / 256, D_MODEL), 128, SM_SIZE, stream>>>(
      Kh, u_h, u, Dp, out);
}
